// ArcticMoE_35313221108249
// MI455X (gfx1250) — compile-verified, matrix operand order checked by reading, not run on hardware
//
#include <hip/hip_runtime.h>

// ---------------------------------------------------------------------------
// ArcticMoE for MI455X (gfx1250): bf16 WMMA grouped-GEMM MoE FFN.
//   hidden_states (2048,1024) f32, gate_w (8,1024) f32,
//   w1 (8,1024,2048) f32, w2 (8,2048,1024) f32, top-2 of 8 experts.
// Pipeline: gate -> FFN1 (silu(X@w1) -> H bf16) -> FFN2 (H@w2 -> atomic out).
// ---------------------------------------------------------------------------

typedef __attribute__((ext_vector_type(16))) __bf16 v16bf;
typedef __attribute__((ext_vector_type(8)))  __bf16 v8bf;
typedef __attribute__((ext_vector_type(4)))  __bf16 v4bf;
typedef __attribute__((ext_vector_type(8)))  float  v8f;

constexpr int T_TOK = 2048;   // tokens
constexpr int DDIM  = 1024;   // hidden dim
constexpr int IDIM  = 2048;   // intermediate dim
constexpr int NEXP  = 8;      // experts

// GEMM tiling: 128x128 block tile, K-step 32 (one bf16 WMMA K), 8 waves,
// each wave owns a 32x64 sub-tile = 2x4 WMMA accumulators.
constexpr int BM  = 128;
constexpr int BN  = 128;
constexpr int BK  = 32;
constexpr int LDT = 40;       // LDS tile stride in halves (80 B = 5*16 B -> b128 aligned)

// ---------------------------------------------------------------------------
// gfx1250 async global->LDS copy path (ASYNCcnt-tracked).  The builtin takes
// 16-byte int-vector pointers; global side is addrspace(1), LDS side
// addrspace(3) (same cast pattern HIP headers use for AS-qualified builtins).
// ---------------------------------------------------------------------------
#if defined(__has_builtin)
#if __has_builtin(__builtin_amdgcn_global_load_async_to_lds_b128) && \
    __has_builtin(__builtin_amdgcn_s_wait_asynccnt)
#define MOE_ASYNC_LDS 1
#endif
#endif
#ifndef MOE_ASYNC_LDS
#define MOE_ASYNC_LDS 0
#endif

typedef int mv4i __attribute__((vector_size(16)));
typedef __attribute__((address_space(1))) mv4i glb_v4i;
typedef __attribute__((address_space(3))) mv4i lds_v4i;

__device__ __forceinline__ glb_v4i* as_glb4(const void* p) {
  return (glb_v4i*)p;                     // generic -> global addrspacecast
}
__device__ __forceinline__ lds_v4i* as_lds4(const void* p) {
  return (lds_v4i*)p;                     // generic -> LDS addrspacecast
}

// ---------------------------------------------------------------------------
// Kernel 1: gating.  One wave per token: logits = hs[t] . gate_w[e],
// softmax, top-2, append (pair_row, score) into per-expert buckets.
// ---------------------------------------------------------------------------
__global__ __launch_bounds__(256) void moe_gate(
    const float* __restrict__ hs, const float* __restrict__ gw,
    int* __restrict__ cnt, int* __restrict__ prs, float* __restrict__ scs)
{
  __shared__ float g[NEXP * DDIM];              // 32 KB of the 320 KB WGP LDS
#if MOE_ASYNC_LDS
  for (int i = threadIdx.x * 4; i < NEXP * DDIM; i += 256 * 4)
    __builtin_amdgcn_global_load_async_to_lds_b128(as_glb4(gw + i), as_lds4(&g[i]), 0, 0);
  __builtin_amdgcn_s_wait_asynccnt(0);
#else
  for (int i = threadIdx.x; i < NEXP * DDIM; i += 256) g[i] = gw[i];
#endif
  __syncthreads();

  const int lane = threadIdx.x & 31;
  const int wave = threadIdx.x >> 5;
  const int t    = blockIdx.x * 8 + wave;

  float acc[NEXP];
#pragma unroll
  for (int e = 0; e < NEXP; ++e) acc[e] = 0.f;

  const float* hrow = hs + (size_t)t * DDIM;
  for (int d = lane; d < DDIM; d += 32) {       // wave32: coalesced 128B/iter
    const float h = hrow[d];
#pragma unroll
    for (int e = 0; e < NEXP; ++e) acc[e] = fmaf(h, g[e * DDIM + d], acc[e]);
  }
#pragma unroll
  for (int e = 0; e < NEXP; ++e) {
#pragma unroll
    for (int off = 16; off > 0; off >>= 1)
      acc[e] += __shfl_xor(acc[e], off, 32);
  }

  if (lane == 0) {
    float mx = acc[0];
#pragma unroll
    for (int e = 1; e < NEXP; ++e) mx = fmaxf(mx, acc[e]);
    float p[NEXP], sum = 0.f;
#pragma unroll
    for (int e = 0; e < NEXP; ++e) { p[e] = __expf(acc[e] - mx); sum += p[e]; }
    const float inv = 1.f / sum;
#pragma unroll
    for (int e = 0; e < NEXP; ++e) p[e] *= inv;

    int e1 = 0; float p1 = p[0];
#pragma unroll
    for (int e = 1; e < NEXP; ++e) if (p[e] > p1) { p1 = p[e]; e1 = e; }
    int e2 = -1; float p2 = -1.f;
#pragma unroll
    for (int e = 0; e < NEXP; ++e) if (e != e1 && p[e] > p2) { p2 = p[e]; e2 = e; }

    int s1 = atomicAdd(&cnt[e1], 1);
    prs[e1 * T_TOK + s1] = t * 2;     scs[e1 * T_TOK + s1] = p1;
    int s2 = atomicAdd(&cnt[e2], 1);
    prs[e2 * T_TOK + s2] = t * 2 + 1; scs[e2 * T_TOK + s2] = p2;
  }
}

// ---------------------------------------------------------------------------
// bf16 WMMA fragment fetch from LDS, following the CDNA5 16-bit A/B layout:
//   lanes 0-15 : K 0..7 in v[0..3],  K 16..23 in v[4..7]
//   lanes 16-31: K 8..15 in v[0..3], K 24..31 in v[4..7]
// Two 16-byte LDS reads per fragment.
// ---------------------------------------------------------------------------
__device__ __forceinline__ v16bf frag_ld(const __bf16* base) {
  union { v16bf v; v8bf h[2]; } u;
  u.h[0] = *(const v8bf*)(base);
  u.h[1] = *(const v8bf*)(base + 16);
  return u.v;
}

// ---------------------------------------------------------------------------
// Kernel 2: FFN1  H[pair_row] = silu( gather(hs) @ w1[e] ), H stored bf16.
// grid = (IDIM/BN, T/BM, NEXP), block = 256 (8 waves).
// ---------------------------------------------------------------------------
__global__ __launch_bounds__(256) void moe_ffn1(
    const float* __restrict__ hs, const float* __restrict__ w1,
    const int* __restrict__ cnt, const int* __restrict__ prs,
    __bf16* __restrict__ H)
{
  const int e  = blockIdx.z;
  const int mt = blockIdx.y;
  const int nt = blockIdx.x;
  const int count = cnt[e];
  if (mt * BM >= count) return;                 // dynamic grouped-GEMM exit

  __shared__ __bf16 As[BM * LDT];               // 10 KB
  __shared__ __bf16 Bs[BN * LDT];               // 10 KB (stored N-major)

  const int tid  = threadIdx.x;
  const int lane = tid & 31, wave = tid >> 5;
  const int wm = wave >> 1, wn = wave & 1;      // 4x2 wave grid over 128x128
  const int l16 = lane & 15, lh = lane >> 4;

  // A-tile loaders: thread -> (row = tid/8 + 32p, k4 = (tid%8)*4), float4 each.
  const int a_kk   = (tid & 7) * 4;
  const int a_row0 = tid >> 3;
  const float* arow[4];
#pragma unroll
  for (int p = 0; p < 4; ++p) {
    int s  = mt * BM + a_row0 + p * 32;
    int sc = s < count ? s : count - 1;         // clamp; garbage rows predicated later
    int tok = prs[e * T_TOK + sc] >> 1;
    arow[p] = hs + (size_t)tok * DDIM;
  }
  // B-tile loaders: thread -> (n4 = (tid%32)*4, k = tid/32 + 8p), float4 over N.
  const int b_n4 = (tid & 31) * 4;
  const int b_kd = tid >> 5;
  const float* w1e = w1 + (size_t)e * DDIM * IDIM + (size_t)nt * BN;

  v8f acc[2][4];
#pragma unroll
  for (int r = 0; r < 2; ++r)
#pragma unroll
    for (int c = 0; c < 4; ++c)
      acc[r][c] = (v8f){0.f, 0.f, 0.f, 0.f, 0.f, 0.f, 0.f, 0.f};

  for (int k0 = 0; k0 < DDIM; k0 += BK) {
    // --- stage A (f32 -> bf16) ---
#pragma unroll
    for (int p = 0; p < 4; ++p) {
      const float4 v = *(const float4*)(arow[p] + k0 + a_kk);
      v4bf b; b[0] = (__bf16)v.x; b[1] = (__bf16)v.y; b[2] = (__bf16)v.z; b[3] = (__bf16)v.w;
      *(v4bf*)(&As[(a_row0 + p * 32) * LDT + a_kk]) = b;
    }
    // --- stage B transposed (f32 row-major K x N -> LDS N-major) ---
#pragma unroll
    for (int p = 0; p < 4; ++p) {
      const int k = b_kd + p * 8;
      const float4 v = *(const float4*)(w1e + (size_t)(k0 + k) * IDIM + b_n4);
      Bs[(b_n4 + 0) * LDT + k] = (__bf16)v.x;
      Bs[(b_n4 + 1) * LDT + k] = (__bf16)v.y;
      Bs[(b_n4 + 2) * LDT + k] = (__bf16)v.z;
      Bs[(b_n4 + 3) * LDT + k] = (__bf16)v.w;
    }
    if (k0 + BK < DDIM)                          // warm L2/WGP$ for next K tile
      __builtin_prefetch(w1e + (size_t)(k0 + BK + b_kd) * IDIM + b_n4, 0, 1);
    __syncthreads();

    v16bf a[2], b[4];
#pragma unroll
    for (int r = 0; r < 2; ++r)
      a[r] = frag_ld(&As[(wm * 32 + r * 16 + l16) * LDT + lh * 8]);
#pragma unroll
    for (int c = 0; c < 4; ++c)
      b[c] = frag_ld(&Bs[(wn * 64 + c * 16 + l16) * LDT + lh * 8]);

#pragma unroll
    for (int r = 0; r < 2; ++r)
#pragma unroll
      for (int c = 0; c < 4; ++c)
        acc[r][c] = __builtin_amdgcn_wmma_f32_16x16x32_bf16(
            false, a[r], false, b[c], (short)0, acc[r][c], false, false);
    __syncthreads();
  }

  // Epilogue: silu, convert to bf16, scatter to H[pair_row].
#pragma unroll
  for (int r = 0; r < 2; ++r) {
#pragma unroll
    for (int j = 0; j < 8; ++j) {
      const int m_local = wm * 32 + r * 16 + lh * 8 + j;   // C layout: M = j + 8*lh
      const int s = mt * BM + m_local;
      if (s < count) {
        const int pr = prs[e * T_TOK + s];
        __bf16* hrow = H + (size_t)pr * IDIM + (size_t)nt * BN;
#pragma unroll
        for (int c = 0; c < 4; ++c) {
          const float x = acc[r][c][j];
          hrow[wn * 64 + c * 16 + l16] = (__bf16)(x / (1.f + __expf(-x)));
        }
      }
    }
  }
}

// ---------------------------------------------------------------------------
// Kernel 3: FFN2  out[token] += score * ( gather(H) @ w2[e] ).
// grid = (DDIM/BN, T/BM, NEXP), block = 256.
// A tile is pure bf16 byte-movement -> async global->LDS unit (ASYNCcnt),
// freeing VALU/DS slots for the B-tile f32->bf16 conversion stream.
// ---------------------------------------------------------------------------
__global__ __launch_bounds__(256) void moe_ffn2(
    const __bf16* __restrict__ H, const float* __restrict__ w2,
    const int* __restrict__ cnt, const int* __restrict__ prs,
    const float* __restrict__ scs, float* __restrict__ out)
{
  const int e  = blockIdx.z;
  const int mt = blockIdx.y;
  const int nt = blockIdx.x;
  const int count = cnt[e];
  if (mt * BM >= count) return;

  __shared__ __bf16 As[BM * LDT];
  __shared__ __bf16 Bs[BN * LDT];

  const int tid  = threadIdx.x;
  const int lane = tid & 31, wave = tid >> 5;
  const int wm = wave >> 1, wn = wave & 1;
  const int l16 = lane & 15, lh = lane >> 4;

  // A loaders: H is already bf16 -> 16 B per thread, rows gathered by pair_row.
  const int a_kk   = (tid & 3) * 8;             // half offset, 16 B aligned
  const int a_row0 = tid >> 2;                  // 0..63, 2 passes
  const __bf16* arow[2];
#pragma unroll
  for (int p = 0; p < 2; ++p) {
    int s  = mt * BM + a_row0 + p * 64;
    int sc = s < count ? s : count - 1;
    arow[p] = H + (size_t)prs[e * T_TOK + sc] * IDIM;
  }
  const int b_n4 = (tid & 31) * 4;
  const int b_kd = tid >> 5;
  const float* w2e = w2 + (size_t)e * IDIM * DDIM + (size_t)nt * BN;

  v8f acc[2][4];
#pragma unroll
  for (int r = 0; r < 2; ++r)
#pragma unroll
    for (int c = 0; c < 4; ++c)
      acc[r][c] = (v8f){0.f, 0.f, 0.f, 0.f, 0.f, 0.f, 0.f, 0.f};

  for (int k0 = 0; k0 < IDIM; k0 += BK) {
#pragma unroll
    for (int p = 0; p < 2; ++p) {               // 16 B/lane global->LDS copies
#if MOE_ASYNC_LDS
      __builtin_amdgcn_global_load_async_to_lds_b128(
          as_glb4(arow[p] + k0 + a_kk),
          as_lds4(&As[(a_row0 + p * 64) * LDT + a_kk]), 0, 0);
#else
      const uint4 v = *(const uint4*)(arow[p] + k0 + a_kk);
      *(uint4*)(&As[(a_row0 + p * 64) * LDT + a_kk]) = v;
#endif
    }
#pragma unroll
    for (int p = 0; p < 4; ++p) {
      const int k = b_kd + p * 8;
      const float4 v = *(const float4*)(w2e + (size_t)(k0 + k) * DDIM + b_n4);
      Bs[(b_n4 + 0) * LDT + k] = (__bf16)v.x;
      Bs[(b_n4 + 1) * LDT + k] = (__bf16)v.y;
      Bs[(b_n4 + 2) * LDT + k] = (__bf16)v.z;
      Bs[(b_n4 + 3) * LDT + k] = (__bf16)v.w;
    }
    if (k0 + BK < IDIM)
      __builtin_prefetch(w2e + (size_t)(k0 + BK + b_kd) * DDIM + b_n4, 0, 1);
#if MOE_ASYNC_LDS
    __builtin_amdgcn_s_wait_asynccnt(0);        // our wave's async copies landed
#endif
    __syncthreads();

    v16bf a[2], b[4];
#pragma unroll
    for (int r = 0; r < 2; ++r)
      a[r] = frag_ld(&As[(wm * 32 + r * 16 + l16) * LDT + lh * 8]);
#pragma unroll
    for (int c = 0; c < 4; ++c)
      b[c] = frag_ld(&Bs[(wn * 64 + c * 16 + l16) * LDT + lh * 8]);

#pragma unroll
    for (int r = 0; r < 2; ++r)
#pragma unroll
      for (int c = 0; c < 4; ++c)
        acc[r][c] = __builtin_amdgcn_wmma_f32_16x16x32_bf16(
            false, a[r], false, b[c], (short)0, acc[r][c], false, false);
    __syncthreads();
  }

  // Epilogue: scale by routing score, atomic-accumulate into out[token].
#pragma unroll
  for (int r = 0; r < 2; ++r) {
#pragma unroll
    for (int j = 0; j < 8; ++j) {
      const int m_local = wm * 32 + r * 16 + lh * 8 + j;
      const int s = mt * BM + m_local;
      if (s < count) {
        const int   pr = prs[e * T_TOK + s];
        const float sv = scs[e * T_TOK + s];
        float* orow = out + (size_t)(pr >> 1) * DDIM + (size_t)nt * BN;
#pragma unroll
        for (int c = 0; c < 4; ++c)
          unsafeAtomicAdd(&orow[wn * 64 + c * 16 + l16], sv * acc[r][c][j]);
      }
    }
  }
}

// ---------------------------------------------------------------------------
// Host launcher.
// ---------------------------------------------------------------------------
extern "C" void kernel_launch(void* const* d_in, const int* in_sizes, int n_in,
                              void* d_out, int out_size, void* d_ws, size_t ws_size,
                              hipStream_t stream) {
  (void)in_sizes; (void)n_in; (void)out_size; (void)ws_size;
  const float* hs = (const float*)d_in[0];
  const float* gw = (const float*)d_in[1];
  const float* w1 = (const float*)d_in[2];
  const float* w2 = (const float*)d_in[3];
  float* out = (float*)d_out;

  // Workspace layout.
  char* ws = (char*)d_ws;
  int*    cnt = (int*)ws;                                       // 8 ints (pad 256)
  int*    prs = (int*)  (ws + 256);                             // E*T pair rows
  float*  scs = (float*)(ws + 256 + sizeof(int)   * NEXP * T_TOK);
  size_t  hof = 256 + (sizeof(int) + sizeof(float)) * (size_t)NEXP * T_TOK;
  hof = (hof + 255) & ~(size_t)255;
  __bf16* H = (__bf16*)(ws + hof);                              // 2*T*I bf16 = 16 MB

  (void)hipMemsetAsync(d_out, 0, (size_t)T_TOK * DDIM * sizeof(float), stream);
  (void)hipMemsetAsync(d_ws, 0, 256, stream);

  moe_gate<<<T_TOK / 8, 256, 0, stream>>>(hs, gw, cnt, prs, scs);
  moe_ffn1<<<dim3(IDIM / BN, T_TOK / BM, NEXP), 256, 0, stream>>>(hs, w1, cnt, prs, H);
  moe_ffn2<<<dim3(DDIM / BN, T_TOK / BM, NEXP), 256, 0, stream>>>(H, w2, cnt, prs, scs, out);
}
